// GRU_OneLayer_78769700209277
// MI455X (gfx1250) — matrix operand order repaired and checked by reading, not compiled, not run on hardware
//
#include <hip/hip_runtime.h>
#include <hip/hip_bf16.h>
#include <stdint.h>

// ---------------------------------------------------------------------------
// GRU one-layer, L=8192, E=H=2048, gates (r,z,n).
// Phase 1: GI = X * W_ih^T + b_ih  via bf16 WMMA (v_wmma_f32_16x16x32_bf16),
//          4 accumulators (hazard-free interleave) + 2-deep software pipeline
//          on the fragment buffers (partial s_wait_loadcnt instead of 0x0).
// Phase 2: persistent-grid sequential recurrence, W_hh streamed as bf16 (25MB,
//          resident in the 192MB L2); one atomic grid barrier per timestep.
// Phase 3: out = sigmoid(h_T . fc_w + fc_b)  (block 0)
// ---------------------------------------------------------------------------

#define EDIM 2048
#define HDIM 2048
#define GDIM 6144      // 3*HDIM
#define LSEQ 8192
#define NBLK 48        // persistent blocks; 48*128 = 6144 rows

typedef __attribute__((ext_vector_type(8)))  float  v8f;
typedef __attribute__((ext_vector_type(16))) __bf16 v16bf;

union FragU { uint4 q[2]; v16bf v; };

struct Frags { FragU a0, a1, b00, b01, b10, b11; };

__device__ __forceinline__ float uasf(unsigned u) { return __uint_as_float(u); }

__device__ __forceinline__ unsigned short f32_to_bf16_rne(float f) {
    unsigned u = __float_as_uint(f);
    unsigned r = u + 0x7fffu + ((u >> 16) & 1u);
    return (unsigned short)(r >> 16);
}

// -------------------------- f32 -> bf16 convert ----------------------------
__global__ __launch_bounds__(256) void cvt_bf16_kernel(const float* __restrict__ in,
                                                       unsigned short* __restrict__ out,
                                                       int n) {
    int i = blockIdx.x * 256 + threadIdx.x;
    if (i < n) out[i] = f32_to_bf16_rne(in[i]);
}

// -------------------------- zero sync counters -----------------------------
__global__ void init_sync_kernel(unsigned* sync) {
    if (threadIdx.x < 8) sync[threadIdx.x] = 0u;
}

// ----------------------- GEMM fragment helpers -----------------------------
__device__ __forceinline__ void load_frags(Frags& f, const unsigned short* pa,
                                           const unsigned short* pb0,
                                           const unsigned short* pb1, int k) {
    f.a0.q[0]  = *(const uint4*)(pa + k);
    f.a0.q[1]  = *(const uint4*)(pa + k + 8);
    f.a1.q[0]  = *(const uint4*)(pa + k + 32);
    f.a1.q[1]  = *(const uint4*)(pa + k + 40);
    f.b00.q[0] = *(const uint4*)(pb0 + k);
    f.b00.q[1] = *(const uint4*)(pb0 + k + 8);
    f.b01.q[0] = *(const uint4*)(pb0 + k + 32);
    f.b01.q[1] = *(const uint4*)(pb0 + k + 40);
    f.b10.q[0] = *(const uint4*)(pb1 + k);
    f.b10.q[1] = *(const uint4*)(pb1 + k + 8);
    f.b11.q[0] = *(const uint4*)(pb1 + k + 32);
    f.b11.q[1] = *(const uint4*)(pb1 + k + 40);
}

__device__ __forceinline__ void wmma4(const Frags& f, v8f& c0a, v8f& c1a,
                                      v8f& c0b, v8f& c1b) {
    // interleave so each accumulator's next use is 3 WMMAs away (no hazard nops)
    c0a = __builtin_amdgcn_wmma_f32_16x16x32_bf16(false, f.a0.v, false, f.b00.v,
                                                  (short)0, c0a, false, false);
    c1a = __builtin_amdgcn_wmma_f32_16x16x32_bf16(false, f.a0.v, false, f.b10.v,
                                                  (short)0, c1a, false, false);
    c0b = __builtin_amdgcn_wmma_f32_16x16x32_bf16(false, f.a1.v, false, f.b01.v,
                                                  (short)0, c0b, false, false);
    c1b = __builtin_amdgcn_wmma_f32_16x16x32_bf16(false, f.a1.v, false, f.b11.v,
                                                  (short)0, c1b, false, false);
}

// ---------------------------------------------------------------------------
// Phase 1: GI[t,g] = sum_e Xbf[t,e]*Wihbf[g,e] + b_ih[g]
// Block: 256 thr = 8 waves. Block tile: 32 (t) x 128 (g).
// Wave tile: 16 (t) x 32 (g). K pipelined 2-deep in 64-wide chunks.
// grid = (GDIM/128, LSEQ/32) = (48, 256)
// ---------------------------------------------------------------------------
__global__ __launch_bounds__(256) void gemm_gi_kernel(const unsigned short* __restrict__ Xbf,
                                                      const unsigned short* __restrict__ Wbf,
                                                      const float* __restrict__ b_ih,
                                                      float* __restrict__ GI) {
    const int lane  = threadIdx.x & 31;
    const int wave  = threadIdx.x >> 5;
    const int half  = lane >> 4;   // 0 or 1
    const int idx16 = lane & 15;

    const int g0 = blockIdx.x * 128 + (wave & 3) * 32;
    const int t0 = blockIdx.y * 32  + (wave >> 2) * 16;

    // A fragment (16x32, 16-bit): lane -> row = lane&15, K = half*16 + {0..15}
    const unsigned short* pa  = Xbf + (size_t)(t0 + idx16) * EDIM + half * 16;
    // B fragment (32x16, 16-bit): lane -> col = lane&15, K = half*16 + {0..15}
    const unsigned short* pb0 = Wbf + (size_t)(g0 + idx16) * EDIM + half * 16;
    const unsigned short* pb1 = pb0 + (size_t)16 * EDIM;

    v8f c0a = {}; v8f c0b = {}; v8f c1a = {}; v8f c1b = {};

    Frags fa, fb;
    load_frags(fa, pa, pb0, pb1, 0);
    load_frags(fb, pa, pb0, pb1, 64);

    int k = 0;
    #pragma unroll 1
    for (; k + 128 < EDIM; k += 128) {
        wmma4(fa, c0a, c1a, c0b, c1b);
        load_frags(fa, pa, pb0, pb1, k + 128);
        wmma4(fb, c0a, c1a, c0b, c1b);
        load_frags(fb, pa, pb0, pb1, k + 192);
    }
    // epilogue: fa holds k=EDIM-128, fb holds k=EDIM-64
    wmma4(fa, c0a, c1a, c0b, c1b);
    wmma4(fb, c0a, c1a, c0b, c1b);

    v8f c0 = c0a + c0b;
    v8f c1 = c1a + c1b;

    // C/D layout: lane L -> col = L&15 ; VGPR v -> row = v + 8*(L>>4)
    const int col = idx16;
    const float bia0 = b_ih[g0 + col];
    const float bia1 = b_ih[g0 + 16 + col];
    #pragma unroll
    for (int v = 0; v < 8; ++v) {
        const int t = t0 + v + 8 * half;
        GI[(size_t)t * GDIM + g0 + col]      = c0[v] + bia0;
        GI[(size_t)t * GDIM + g0 + 16 + col] = c1[v] + bia1;
    }
}

// ------------------------- grid barrier (persistent) -----------------------
__device__ __forceinline__ void grid_barrier(unsigned* cnt, unsigned* gen, int nblk) {
    __syncthreads();
    if (threadIdx.x == 0) {
        __threadfence();
        unsigned g = __hip_atomic_load(gen, __ATOMIC_RELAXED, __HIP_MEMORY_SCOPE_AGENT);
        unsigned a = __hip_atomic_fetch_add(cnt, 1u, __ATOMIC_ACQ_REL, __HIP_MEMORY_SCOPE_AGENT);
        if (a == (unsigned)nblk - 1u) {
            __hip_atomic_store(cnt, 0u, __ATOMIC_RELAXED, __HIP_MEMORY_SCOPE_AGENT);
            __hip_atomic_fetch_add(gen, 1u, __ATOMIC_RELEASE, __HIP_MEMORY_SCOPE_AGENT);
        } else {
            while (__hip_atomic_load(gen, __ATOMIC_ACQUIRE, __HIP_MEMORY_SCOPE_AGENT) == g) {
                __builtin_amdgcn_s_sleep(1);
            }
        }
        __threadfence();
    }
    __syncthreads();
}

__device__ __forceinline__ float sigmoidf_fast(float x) {
    return 1.0f / (1.0f + __expf(-x));
}

// ---------------------------------------------------------------------------
// Phase 2+3: persistent sequential GRU.
// 48 blocks x 256 threads (8 waves). Each block owns 128 rows of gh per step
// (wave -> 16 rows; lane -> contiguous K-slice of 64, h cached in registers).
// One grid barrier per step; gh double-buffered on t parity; every block
// redundantly computes the 2048-wide gate combine into its private LDS h.
// GI row for the current step is prefetched during the GEMV so the combine
// after the barrier hits warm cache lines.
// ---------------------------------------------------------------------------
__global__ __launch_bounds__(256) void gru_seq_kernel(const unsigned short* __restrict__ Whh,
                                                      const float* __restrict__ GI,
                                                      const float* __restrict__ b_hh,
                                                      const float* __restrict__ fc_w,
                                                      const float* __restrict__ fc_b,
                                                      float* __restrict__ ghbuf,  // [2][GDIM]
                                                      unsigned* __restrict__ sync,
                                                      float* __restrict__ out) {
    __shared__ float hsh[HDIM];     // 8 KB: block-private copy of h (identical in all blocks)
    __shared__ float red[256];

    const int tid  = threadIdx.x;
    const int lane = tid & 31;
    const int wave = tid >> 5;
    const int rowBase = blockIdx.x * 128 + wave * 16;

    unsigned* cnt = sync;
    unsigned* gen = sync + 1;

    for (int i = tid; i < HDIM; i += 256) hsh[i] = 0.0f;
    __syncthreads();

    for (int t = 0; t < LSEQ; ++t) {
        const float* git = GI + (size_t)t * GDIM;
        // warm this step's GI row (6144 f32 = 24KB) while the GEMV runs
        __builtin_prefetch((const void*)(git + tid * 24), 0, 1);

        // ---- cache this lane's K-slice of h: 64 contiguous floats ----
        float hbuf[64];
        {
            const float4* h4 = (const float4*)hsh;
            #pragma unroll
            for (int q = 0; q < 16; ++q) {
                float4 v = h4[lane * 16 + q];
                hbuf[4 * q + 0] = v.x; hbuf[4 * q + 1] = v.y;
                hbuf[4 * q + 2] = v.z; hbuf[4 * q + 3] = v.w;
            }
        }

        // ---- GEMV: 16 rows per wave, partial dot over this lane's slice ----
        float acc[16];
        #pragma unroll
        for (int r = 0; r < 16; ++r) {
            const uint4* wp = (const uint4*)(Whh + (size_t)(rowBase + r) * HDIM + lane * 64);
            if (r < 15) __builtin_prefetch((const void*)(wp + (HDIM / 8)), 0, 1);
            float a = 0.0f;
            #pragma unroll
            for (int j = 0; j < 8; ++j) {
                uint4 w = wp[j];
                a += uasf(w.x << 16)         * hbuf[8 * j + 0];
                a += uasf(w.x & 0xffff0000u) * hbuf[8 * j + 1];
                a += uasf(w.y << 16)         * hbuf[8 * j + 2];
                a += uasf(w.y & 0xffff0000u) * hbuf[8 * j + 3];
                a += uasf(w.z << 16)         * hbuf[8 * j + 4];
                a += uasf(w.z & 0xffff0000u) * hbuf[8 * j + 5];
                a += uasf(w.w << 16)         * hbuf[8 * j + 6];
                a += uasf(w.w & 0xffff0000u) * hbuf[8 * j + 7];
            }
            acc[r] = a;
        }

        // ---- reduce each row across the 32 lanes ----
        #pragma unroll
        for (int r = 0; r < 16; ++r) {
            float a = acc[r];
            #pragma unroll
            for (int s = 16; s > 0; s >>= 1) a += __shfl_xor(a, s, 32);
            acc[r] = a;
        }

        float* ghw = ghbuf + (size_t)(t & 1) * GDIM;
        if (lane == 0) {
            #pragma unroll
            for (int r = 0; r < 16; ++r) ghw[rowBase + r] = acc[r];
        }

        grid_barrier(cnt, gen, NBLK);

        // ---- redundant full combine into block-private h (no 2nd barrier) ----
        const float* ghr = ghbuf + (size_t)(t & 1) * GDIM;
        for (int j = tid; j < HDIM; j += 256) {
            float hr = ghr[j]            + b_hh[j];
            float hz = ghr[j + HDIM]     + b_hh[j + HDIM];
            float hn = ghr[j + 2 * HDIM] + b_hh[j + 2 * HDIM];
            float ir  = git[j];
            float iz  = git[j + HDIM];
            float in_ = git[j + 2 * HDIM];
            float rg = sigmoidf_fast(ir + hr);
            float zg = sigmoidf_fast(iz + hz);
            float ng = tanhf(in_ + rg * hn);
            float hp = hsh[j];
            hsh[j] = (1.0f - zg) * ng + zg * hp;
        }
        __syncthreads();
    }

    // ---- Phase 3: block 0 computes sigmoid(h . fc_w + fc_b) ----
    if (blockIdx.x == 0) {
        float p = 0.0f;
        for (int i = tid; i < HDIM; i += 256) p += hsh[i] * fc_w[i];
        red[tid] = p;
        __syncthreads();
        for (int s = 128; s > 0; s >>= 1) {
            if (tid < s) red[tid] += red[tid + s];
            __syncthreads();
        }
        if (tid == 0) out[0] = sigmoidf_fast(red[0] + fc_b[0]);
    }
}

// ---------------------------------------------------------------------------
// Host-side launcher. Workspace layout (bytes, ~285.3 MB total):
//   Xbf   @ 0          : LSEQ*EDIM*2      = 33,554,432
//   Wihbf @ 33,554,432 : GDIM*EDIM*2      = 25,165,824
//   Whhbf @ 58,720,256 : GDIM*HDIM*2      = 25,165,824
//   GI    @ 83,886,080 : LSEQ*GDIM*4      = 201,326,592
//   gh    @ 285,212,672: 2*GDIM*4         = 49,152
//   sync  @ 285,261,824: 256
// ---------------------------------------------------------------------------
extern "C" void kernel_launch(void* const* d_in, const int* in_sizes, int n_in,
                              void* d_out, int out_size, void* d_ws, size_t ws_size,
                              hipStream_t stream) {
    (void)in_sizes; (void)n_in; (void)out_size; (void)ws_size;

    const float* x    = (const float*)d_in[0];  // [1, L, E]
    const float* W_ih = (const float*)d_in[1];  // [3H, E]
    const float* W_hh = (const float*)d_in[2];  // [3H, H]
    const float* b_ih = (const float*)d_in[3];  // [3H]
    const float* b_hh = (const float*)d_in[4];  // [3H]
    const float* fc_w = (const float*)d_in[5];  // [1, H]
    const float* fc_b = (const float*)d_in[6];  // [1]
    float* out = (float*)d_out;

    char* ws = (char*)d_ws;
    unsigned short* Xbf   = (unsigned short*)(ws + 0);
    unsigned short* Wihbf = (unsigned short*)(ws + 33554432);
    unsigned short* Whhbf = (unsigned short*)(ws + 58720256);
    float*          GI    = (float*)(ws + 83886080);
    float*          gh    = (float*)(ws + 285212672);
    unsigned*       sync  = (unsigned*)(ws + 285261824);

    // f32 -> bf16 conversions
    {
        int n = LSEQ * EDIM;
        cvt_bf16_kernel<<<(n + 255) / 256, 256, 0, stream>>>(x, Xbf, n);
    }
    {
        int n = GDIM * EDIM;
        cvt_bf16_kernel<<<(n + 255) / 256, 256, 0, stream>>>(W_ih, Wihbf, n);
    }
    {
        int n = GDIM * HDIM;
        cvt_bf16_kernel<<<(n + 255) / 256, 256, 0, stream>>>(W_hh, Whhbf, n);
    }

    init_sync_kernel<<<1, 32, 0, stream>>>(sync);

    // Phase 1: WMMA GEMM for input-side gate preactivations (+b_ih)
    gemm_gi_kernel<<<dim3(GDIM / 128, LSEQ / 32), 256, 0, stream>>>(Xbf, Wihbf, b_ih, GI);

    // Phase 2+3: persistent sequential recurrence + final score
    gru_seq_kernel<<<NBLK, 256, 0, stream>>>(Whhbf, GI, b_hh, fc_w, fc_b, gh, sync, out);
}
